// CirConv2d_40097814676034
// MI455X (gfx1250) — matrix operand order repaired and checked by reading, not compiled, not run on hardware
//
#include <hip/hip_runtime.h>
#include <hip/hip_bf16.h>

typedef __attribute__((ext_vector_type(16))) _Float16 v16h;
typedef __attribute__((ext_vector_type(8)))  _Float16 v8h;
typedef __attribute__((ext_vector_type(8)))  float    v8f;

#define CIN   256
#define COUT  256
#define HW    56
#define SP    3136          // 56*56
#define NIMG  16
#define NTOT  (NIMG*SP)     // 50176  (N dim of implicit GEMM)
#define KTOT  (9*CIN)       // 2304   (K dim)
#define KSTEPS (KTOT/32)    // 72
#define BLK_N 128
#define LDS_STRIDE 48       // halfs per row: 96B rows -> 32B aligned v16h loads
#define WSTEP (16*32*16)    // halfs of packed A per K-step

// ---------------------------------------------------------------------------
// Kernel 1: softmax(alphas) + block-circulant weight synthesis, scattered
// directly into WMMA A-fragment packed layout:
//   Wp[((ks*16 + mtile)*32 + lane)*16 + h]  (f16)
// K index kk = (kh*3+kw)*256 + c ;  ks = kk>>5, klocal = kk&31.
// 16-bit A 16x32 fragment layout (ISA 7.12.2):
//   lane<16 : M=lane,   halfs 0..7 = K 0..7,  halfs 8..15 = K 16..23
//   lane>=16: M=lane-16,halfs 0..7 = K 8..15, halfs 8..15 = K 24..31
// ---------------------------------------------------------------------------
__global__ void synth_pack_kernel(const float* __restrict__ W,
                                  const float* __restrict__ alphas,
                                  const float* __restrict__ gumbels,
                                  _Float16* __restrict__ Wp) {
    __shared__ float tile[64][65];   // padded to kill bank conflicts
    __shared__ float aw[7];
    const int t = threadIdx.x;

    if (t == 0) {
        float l[7], mx = -1e30f;
        #pragma unroll
        for (int i = 0; i < 7; ++i) {
            l[i] = alphas[i] + 1e-4f * gumbels[i];
            mx = fmaxf(mx, l[i]);
        }
        float s = 0.f;
        #pragma unroll
        for (int i = 0; i < 7; ++i) { l[i] = __expf(l[i] - mx); s += l[i]; }
        #pragma unroll
        for (int i = 0; i < 7; ++i) aw[i] = l[i] / s;
    }

    const int ob  = blockIdx.x;          // O block of 64 (0..3)
    const int ib  = blockIdx.y;          // I block of 64 (0..3)
    const int seg = blockIdx.z;          // kh*3+kw (0..8)
    const int kh  = seg / 3, kw = seg % 3;

    for (int e = t; e < 64 * 64; e += 256) {
        const int bi = e >> 6, bj = e & 63;
        const int o = ob * 64 + bi, c = ib * 64 + bj;
        tile[bi][bj] = W[((size_t)(o * CIN + c) * 3 + kh) * 3 + kw];
    }
    __syncthreads();

    for (int e = t; e < 64 * 64; e += 256) {
        const int bi = e >> 6, bj = e & 63;
        float acc = aw[0] * tile[bi][bj];
        #pragma unroll
        for (int lb = 1; lb <= 6; ++lb) {
            const int b = 1 << lb, mask = b - 1;
            const int bi0 = bi & ~mask, bj0 = bj & ~mask;
            const int d = (bj - bi) & mask;
            float s = 0.f;
            for (int r = 0; r < b; ++r)
                s += tile[bi0 + r][bj0 + ((r + d) & mask)];
            acc += aw[lb] * (s * (1.0f / (float)b));
        }
        // scatter into A-fragment packed order
        const int o = ob * 64 + bi, c = ib * 64 + bj;
        const int kk = seg * CIN + c;
        const int ks = kk >> 5, kl = kk & 31;
        const int mt = o >> 4, m = o & 15;
        int lane, h;
        if (kl < 8)       { lane = m;      h = kl;      }
        else if (kl < 16) { lane = m + 16; h = kl - 8;  }
        else if (kl < 24) { lane = m;      h = kl - 8;  }
        else              { lane = m + 16; h = kl - 16; }
        Wp[(((size_t)(ks * 16 + mt) * 32 + lane) << 4) + h] = (_Float16)acc;
    }
}

// ---------------------------------------------------------------------------
// Implicit-GEMM conv:  out[M=o, N=(img,y,x)] += A[M,K] * B[K,N]
// Block tile 128(M) x 128(N), 8 wave32s, each wave 32x64 = 2x4 WMMA tiles.
// Per K-step:  commit staged regs -> Bt[buf] ; ONE barrier ; stage gathers
// for next step ; 8x v_wmma_f32_16x16x32_f16.
// Loop is a MANUAL 2-step body (constant buffer addresses) with unrolling
// disabled so the steady-state code stays I$-resident.
// ---------------------------------------------------------------------------
struct Frag16 { v8h lo, hi; };

__device__ __forceinline__ Frag16 gload_tile(const float* __restrict__ xb,
                                             int ks, int xx, int yy, int kgroup) {
    const int seg = ks >> 3;            // (kh*3+kw), 0..8
    const int kh = seg / 3, kw = seg - kh * 3;
    const int c0 = (ks & 7) << 5;       // channel base of this K step
    const int ys = yy + kh - 1;
    const int xs = xx + kw - 1;
    const bool inb = ((unsigned)ys < (unsigned)HW) & ((unsigned)xs < (unsigned)HW);
    const float* base = xb + (size_t)(c0 + kgroup * 16) * SP + (ys * HW + xs);
    Frag16 f;
    #pragma unroll
    for (int i = 0; i < 8; ++i)         // 128 threads share k -> coalesced
        f.lo[i] = (_Float16)(inb ? base[(size_t)i * SP] : 0.f);
    #pragma unroll
    for (int i = 0; i < 8; ++i)
        f.hi[i] = (_Float16)(inb ? base[(size_t)(i + 8) * SP] : 0.f);
    return f;
}

__device__ __forceinline__ void commit_stage(const Frag16& f,
                                             _Float16 (*Bt)[LDS_STRIDE],
                                             int n, int kgroup) {
    *(v8h*)(&Bt[n][kgroup * 8])      = f.lo;   // two 16B ds_store_b128
    *(v8h*)(&Bt[n][kgroup * 8 + 16]) = f.hi;
}

__device__ __forceinline__ void wmma_step(const _Float16* __restrict__ wb,
                                          const _Float16* brow,
                                          v8f acc[2][4]) {
    const v16h a0 = *(const v16h*)(wb);
    const v16h a1 = *(const v16h*)(wb + 32 * 16);
    #pragma unroll
    for (int jt = 0; jt < 4; ++jt) {
        const v16h b = *(const v16h*)(brow + jt * 16 * LDS_STRIDE);
        acc[0][jt] = __builtin_amdgcn_wmma_f32_16x16x32_f16(
            false, a0, false, b, (short)0, acc[0][jt], false, false);
        acc[1][jt] = __builtin_amdgcn_wmma_f32_16x16x32_f16(
            false, a1, false, b, (short)0, acc[1][jt], false, false);
    }
}

__global__ void __launch_bounds__(256)
conv_wmma_kernel(const float* __restrict__ X,
                 const _Float16* __restrict__ Wp,
                 float* __restrict__ Out) {
    __shared__ _Float16 Bt[2][BLK_N][LDS_STRIDE];

    const int t    = threadIdx.x;
    const int lane = t & 31;
    const int wave = t >> 5;
    const int wm   = wave & 3;              // 4 waves across M (32 rows each)
    const int wn   = wave >> 2;             // 2 waves across N (64 cols each)
    const int s0   = blockIdx.x * BLK_N;    // 392 blocks cover N exactly
    const int mt0  = blockIdx.y * 8 + wm * 2;

    // ---- per-thread loop-invariant fill coordinates (hoisted) ----
    const int n      = t & 127;             // spatial column within tile
    const int kgroup = t >> 7;              // 0..1 -> k = kgroup*16 .. +15
    const int sf     = s0 + n;
    const int fxx    = sf % HW;
    const int fyy    = (sf / HW) % HW;
    const int fimg   = sf / SP;
    const float* xb  = X + (size_t)fimg * CIN * SP;

    // hoisted fragment base pointers
    const _Float16* wb    = Wp + ((size_t)mt0 * 32 + lane) * 16;  // += WSTEP/step
    const _Float16* brow0 = &Bt[0][wn * 64 + (lane & 15)][(lane >> 4) * 16];

    v8f acc[2][4] = {};

    Frag16 stage = gload_tile(xb, 0, fxx, fyy, kgroup);

    #pragma clang loop unroll(disable)
    for (int ks = 0; ks < KSTEPS - 2; ks += 2) {
        // ---- even step: buffer 0 ----
        commit_stage(stage, Bt[0], n, kgroup);
        __syncthreads();
        stage = gload_tile(xb, ks + 1, fxx, fyy, kgroup);
        __builtin_prefetch(wb + WSTEP, 0, 1);           // warm next A block
        __builtin_prefetch(wb + WSTEP + 32 * 16, 0, 1);
        wmma_step(wb, brow0, acc);
        wb += WSTEP;
        // ---- odd step: buffer 1 ----
        commit_stage(stage, Bt[1], n, kgroup);
        __syncthreads();
        stage = gload_tile(xb, ks + 2, fxx, fyy, kgroup);
        __builtin_prefetch(wb + WSTEP, 0, 1);
        __builtin_prefetch(wb + WSTEP + 32 * 16, 0, 1);
        wmma_step(wb, brow0 + BLK_N * LDS_STRIDE, acc);
        wb += WSTEP;
    }
    // ---- step 70 (buffer 0), stage step 71 ----
    commit_stage(stage, Bt[0], n, kgroup);
    __syncthreads();
    stage = gload_tile(xb, KSTEPS - 1, fxx, fyy, kgroup);
    wmma_step(wb, brow0, acc);
    wb += WSTEP;
    // ---- step 71 (buffer 1), final ----
    commit_stage(stage, Bt[1], n, kgroup);
    __syncthreads();
    wmma_step(wb, brow0 + BLK_N * LDS_STRIDE, acc);

    // C/D 16x16 f32 layout: lane<16 -> N=lane,  VGPR r -> M=r
    //                       lane>=16-> N=lane-16,VGPR r -> M=r+8
    const int mofs = (lane < 16) ? 0 : 8;
    #pragma unroll
    for (int jt = 0; jt < 4; ++jt) {
        const int ncol = wn * 64 + jt * 16 + (lane & 15);
        const int s = s0 + ncol;
        const int xx = s % HW;
        const int yy = (s / HW) % HW;
        const int img = s / SP;
        float* obase = Out + (size_t)img * COUT * SP + yy * HW + xx;
        #pragma unroll
        for (int it = 0; it < 2; ++it) {
            const int o0 = (mt0 + it) * 16 + mofs;
            float* op = obase + (size_t)o0 * SP;
            #pragma unroll
            for (int r = 0; r < 8; ++r)
                op[(size_t)r * SP] = acc[it][jt][r];
        }
    }
}

extern "C" void kernel_launch(void* const* d_in, const int* in_sizes, int n_in,
                              void* d_out, int out_size, void* d_ws, size_t ws_size,
                              hipStream_t stream) {
    const float* X       = (const float*)d_in[0];   // [16,256,56,56]
    const float* W       = (const float*)d_in[1];   // [256,256,3,3]
    const float* alphas  = (const float*)d_in[2];   // [7]
    const float* gumbels = (const float*)d_in[3];   // [7]
    float* Out           = (float*)d_out;           // [16,256,56,56]
    _Float16* Wp         = (_Float16*)d_ws;         // packed f16 weight, 1.18 MB

    dim3 g1(4, 4, 9);                  // 64x64 O/I blocks x 9 taps
    synth_pack_kernel<<<g1, 256, 0, stream>>>(W, alphas, gumbels, Wp);

    dim3 g2(NTOT / BLK_N, COUT / 128); // 392 x 2
    conv_wmma_kernel<<<g2, 256, 0, stream>>>(X, Wp, Out);
}
